// BaseWindowAttention_58239756534105
// MI455X (gfx1250) — compile-verified, matrix operand order checked by reading, not run on hardware
//
#include <hip/hip_runtime.h>
#include <hip/hip_bf16.h>

// CDNA5 (gfx1250, wave32) windowed attention.
// Phase 1: qkv = x @ w_qkv          (WMMA f16 GEMM, fp32->f16 LDS staging, f16 out)
// Phase 2: per-(window,head) attn   (async-to-LDS Q/K loads, WMMA QK^T + PV)
// Phase 3: out = attn @ w_out + b   (WMMA f16 GEMM, fp32 out)
// All epilogues stage through LDS so global stores are coalesced b128.
//
// Workspace assumption: ws_size >= 128 MB (96 MB qkv f16 + 32 MB attn f16).

typedef _Float16 half_t;
typedef _Float16 v8h  __attribute__((ext_vector_type(8)));
typedef _Float16 v16h __attribute__((ext_vector_type(16)));
typedef float    v8f  __attribute__((ext_vector_type(8)));

#define WMMA_F16(a, b, c) \
  __builtin_amdgcn_wmma_f32_16x16x32_f16(false, (a), false, (b), (short)0, (c), false, false)

// Async copy 16B global -> LDS (gfx1250 GLOBAL_LOAD_ASYNC_TO_LDS_B128, ASYNCcnt).
__device__ __forceinline__ void async_g2l_b128(void* lds_dst, const void* g_src) {
  const unsigned int       loff = (unsigned int)(unsigned long long)lds_dst;  // flat LDS addr low 32b = offset
  const unsigned long long gptr = (unsigned long long)g_src;
  asm volatile("global_load_async_to_lds_b128 %0, %1, off"
               :: "v"(loff), "v"(gptr)
               : "memory");
}
__device__ __forceinline__ void wait_async0() {
  asm volatile("s_wait_asynccnt 0x0" ::: "memory");
}

// Load a 16x32 f16 WMMA operand fragment from LDS laid out row-major [row][k].
// CDNA5 16-bit A-matrix layout:
//   lanes 0-15 : row = row_base+lane,    K = {kb..kb+7, kb+16..kb+23}, kb = k_off
//   lanes 16-31: row = row_base+lane-16, same but kb = k_off+8
// Two 16B ds_load_b128 per lane.
__device__ __forceinline__ v16h load_frag(const half_t* __restrict__ lds,
                                          int row_base, int k_off, int stride, int lane) {
  const int r  = row_base + (lane & 15);
  const int kb = k_off + ((lane >> 4) << 3);
  const half_t* p = lds + r * stride + kb;
  v8h lo = *(const v8h*)(p);
  v8h hi = *(const v8h*)(p + 16);
  v16h f;
#pragma unroll
  for (int i = 0; i < 8; ++i) { f[i] = lo[i]; f[i + 8] = hi[i]; }
  return f;
}

// ---------------------------------------------------------------------------
// Tiled WMMA GEMM: Out[M,N] = A[M,K] * B[K,N] (+ bias[N])
// Block tile 64(M) x 128(N), K-step 32. 8 waves as 2x4; each wave owns 32x32.
// Epilogue stages through LDS so global stores are b128-coalesced.
// ---------------------------------------------------------------------------
template <bool A_F16, bool ADD_BIAS, bool OUT_F16>
__global__ __launch_bounds__(256) void gemm_wmma_kernel(
    const void* __restrict__ Aptr, const float* __restrict__ B,
    const float* __restrict__ bias, void* __restrict__ Out,
    int M, int N, int K) {
  __shared__ half_t sA[64 * 32];                    // [m][k]
  __shared__ half_t sB[128 * 32];                   // [n][k]  (B staged transposed)
  __shared__ __align__(16) float sOutRaw[64 * 128]; // epilogue staging (f32 or f16 view)

  const int tid  = threadIdx.x;
  const int lane = tid & 31;
  const int wave = tid >> 5;
  const int wm   = wave >> 2;            // 0..1  (M sub-tile)
  const int wn   = wave & 3;             // 0..3  (N sub-tile)
  const int n0   = blockIdx.x * 128;
  const int m0   = blockIdx.y * 64;

  v8f acc00 = {}, acc01 = {}, acc10 = {}, acc11 = {};

  for (int k0 = 0; k0 < K; k0 += 32) {
    // ---- stage A tile (64x32) into LDS as f16 ----
    if (A_F16) {
      const half_t* A = (const half_t*)Aptr;
      const int row = tid >> 2, c8 = (tid & 3) * 8;   // 256 x v8h
      v8h v = *(const v8h*)(A + (size_t)(m0 + row) * K + k0 + c8);
      *(v8h*)(sA + row * 32 + c8) = v;
      if (k0 + 32 < K)
        __builtin_prefetch(A + (size_t)(m0 + row) * K + k0 + 32 + c8, 0, 1);
    } else {
      const float* A = (const float*)Aptr;
#pragma unroll
      for (int it = 0; it < 2; ++it) {
        const int idx = tid + it * 256;               // 512 x float4
        const int row = idx >> 3, c4 = (idx & 7) * 4;
        float4 v = *(const float4*)(A + (size_t)(m0 + row) * K + k0 + c4);
        half_t* d = sA + row * 32 + c4;
        d[0] = (half_t)v.x; d[1] = (half_t)v.y; d[2] = (half_t)v.z; d[3] = (half_t)v.w;
        if (k0 + 32 < K)
          __builtin_prefetch(A + (size_t)(m0 + row) * K + k0 + 32 + c4, 0, 1);
      }
    }
    // ---- stage B tile (32x128) transposed into sB[n][k] as f16 ----
#pragma unroll
    for (int it = 0; it < 4; ++it) {
      const int idx = tid + it * 256;                 // 1024 x float4
      const int kr = idx >> 5, n4 = (idx & 31) * 4;
      float4 v = *(const float4*)(B + (size_t)(k0 + kr) * N + n0 + n4);
      sB[(n4 + 0) * 32 + kr] = (half_t)v.x;
      sB[(n4 + 1) * 32 + kr] = (half_t)v.y;
      sB[(n4 + 2) * 32 + kr] = (half_t)v.z;
      sB[(n4 + 3) * 32 + kr] = (half_t)v.w;
    }
    __syncthreads();

    // ---- 4 WMMA per wave per K-step ----
    v16h a0 = load_frag(sA, wm * 32,      0, 32, lane);
    v16h a1 = load_frag(sA, wm * 32 + 16, 0, 32, lane);
    v16h b0 = load_frag(sB, wn * 32,      0, 32, lane);
    v16h b1 = load_frag(sB, wn * 32 + 16, 0, 32, lane);
    acc00 = WMMA_F16(a0, b0, acc00);
    acc01 = WMMA_F16(a0, b1, acc01);
    acc10 = WMMA_F16(a1, b0, acc10);
    acc11 = WMMA_F16(a1, b1, acc11);
    __syncthreads();
  }

  // ---- epilogue: stage tile in LDS, then coalesced b128 global stores ----
  // C/D layout: VGPR r -> lanes 0-15 M=r, lanes 16-31 M=r+8; N = lane&15.
  const int rsub = (lane >> 4) << 3;
  const int csub = lane & 15;
  const v8f* accs[4] = { &acc00, &acc01, &acc10, &acc11 };

  if (OUT_F16) {
    half_t* sO = (half_t*)sOutRaw;                    // [64][128] f16
#pragma unroll
    for (int i = 0; i < 2; ++i)
#pragma unroll
      for (int j = 0; j < 2; ++j) {
        const v8f a = *accs[i * 2 + j];
        const int lr = wm * 32 + i * 16 + rsub;
        const int lc = wn * 32 + j * 16 + csub;
        const float bv = ADD_BIAS ? bias[n0 + lc] : 0.0f;
#pragma unroll
        for (int r = 0; r < 8; ++r) sO[(lr + r) * 128 + lc] = (half_t)(a[r] + bv);
      }
    __syncthreads();
    half_t* O = (half_t*)Out;
#pragma unroll
    for (int it = 0; it < 4; ++it) {                  // 1024 x v8h
      const int idx = tid + it * 256;
      const int row = idx >> 4, c8 = (idx & 15) * 8;
      *(v8h*)(O + (size_t)(m0 + row) * N + n0 + c8) = *(const v8h*)(sO + row * 128 + c8);
    }
  } else {
    float* sO = sOutRaw;                              // [64][128] f32
#pragma unroll
    for (int i = 0; i < 2; ++i)
#pragma unroll
      for (int j = 0; j < 2; ++j) {
        const v8f a = *accs[i * 2 + j];
        const int lr = wm * 32 + i * 16 + rsub;
        const int lc = wn * 32 + j * 16 + csub;
        const float bv = ADD_BIAS ? bias[n0 + lc] : 0.0f;
#pragma unroll
        for (int r = 0; r < 8; ++r) sO[(lr + r) * 128 + lc] = a[r] + bv;
      }
    __syncthreads();
    float* O = (float*)Out;
#pragma unroll
    for (int it = 0; it < 8; ++it) {                  // 2048 x float4
      const int idx = tid + it * 256;
      const int row = idx >> 5, c4 = (idx & 31) * 4;
      *(float4*)(O + (size_t)(m0 + row) * N + n0 + c4) = *(const float4*)(sO + row * 128 + c4);
    }
  }
}

// ---------------------------------------------------------------------------
// Windowed attention: one block per (bl, head, window). 64 q x 64 k x 64 ch.
// Q/K tiles fetched with GLOBAL_LOAD_ASYNC_TO_LDS_B128 (no VGPR round-trip);
// V goes through VGPRs because it is transposed into LDS.
// ---------------------------------------------------------------------------
__global__ __launch_bounds__(256) void win_attn_kernel(
    const half_t* __restrict__ qkv, const float* __restrict__ pos_emb,
    half_t* __restrict__ outp) {
  __shared__ half_t sQ[64 * 64];    // [q][c]
  __shared__ half_t sK[64 * 64];    // [k][c]
  __shared__ half_t sVT[64 * 64];   // [c][k]  (V transposed)
  __shared__ __align__(16) float sS[64 * 64];  // scores fp32; reused as f16 out staging
  __shared__ half_t sP[64 * 64];    // softmax probs f16
  __shared__ float  sBias[225];     // 15x15 relative position table

  const int tid  = threadIdx.x;
  const int lane = tid & 31;
  const int wave = tid >> 5;

  const int id   = blockIdx.x;
  const int bl   = id >> 9;                 // 0..7  (b*l)
  const int head = (id >> 6) & 7;
  const int win  = id & 63;
  const int nh_i = win >> 3, nw_i = win & 7;

  if (tid < 225) sBias[tid] = pos_emb[tid];

  // ---- async-load Q,K row-major; V transposed via VGPRs ----
  {
    const int lt = tid >> 2;                // window-local token 0..63
    const int c0 = (tid & 3) * 16;          // 16 channels per thread
    const int wr = lt >> 3, wc = lt & 7;
    const int tok = (bl * 64 + nh_i * 8 + wr) * 64 + (nw_i * 8 + wc);
    const size_t base = (size_t)tok * 1536 + head * 64 + c0;

    async_g2l_b128(sQ + lt * 64 + c0,     qkv + base);
    async_g2l_b128(sQ + lt * 64 + c0 + 8, qkv + base + 8);
    async_g2l_b128(sK + lt * 64 + c0,     qkv + base + 512);
    async_g2l_b128(sK + lt * 64 + c0 + 8, qkv + base + 520);

    v8h v0 = *(const v8h*)(qkv + base + 1024);
    v8h v1 = *(const v8h*)(qkv + base + 1032);
#pragma unroll
    for (int i = 0; i < 8; ++i) {
      sVT[(c0 + i) * 64 + lt]     = v0[i];
      sVT[(c0 + 8 + i) * 64 + lt] = v1[i];
    }
    wait_async0();                          // each wave's async copies done
  }
  __syncthreads();

  const int wmq = wave >> 1;   // 0..3 : q-row tile (16 rows)
  const int wnk = wave & 1;    // 0..1 : 32-col half
  const int rsub = (lane >> 4) << 3;
  const int csub = lane & 15;

  // ---- S = Q K^T ----
  {
    v8f s0 = {}, s1 = {};
#pragma unroll
    for (int c0 = 0; c0 < 64; c0 += 32) {
      v16h a  = load_frag(sQ, wmq * 16,      c0, 64, lane);
      v16h b0 = load_frag(sK, wnk * 32,      c0, 64, lane);
      v16h b1 = load_frag(sK, wnk * 32 + 16, c0, 64, lane);
      s0 = WMMA_F16(a, b0, s0);
      s1 = WMMA_F16(a, b1, s1);
    }
#pragma unroll
    for (int r = 0; r < 8; ++r) {
      sS[(wmq * 16 + r + rsub) * 64 + wnk * 32 + csub]      = s0[r];
      sS[(wmq * 16 + r + rsub) * 64 + wnk * 32 + 16 + csub] = s1[r];
    }
  }
  __syncthreads();

  // ---- softmax with relative bias: one thread per query row ----
  if (tid < 64) {
    const int qi = tid >> 3, qj = tid & 7;
    float* row = sS + tid * 64;
    float mx = -3.0e38f;
#pragma unroll 8
    for (int c = 0; c < 64; ++c) {
      const int ki = c >> 3, kj = c & 7;
      const float v = row[c] * 0.125f + sBias[(qi - ki + 7) * 15 + (qj - kj + 7)];
      row[c] = v;
      mx = fmaxf(mx, v);
    }
    float sum = 0.0f;
#pragma unroll 8
    for (int c = 0; c < 64; ++c) { const float e = __expf(row[c] - mx); row[c] = e; sum += e; }
    const float inv = 1.0f / sum;
    half_t* prow = sP + tid * 64;
#pragma unroll 8
    for (int c = 0; c < 64; ++c) prow[c] = (half_t)(row[c] * inv);
  }
  __syncthreads();

  // ---- O = P V  (A = P[q][k], B = sVT[c][k]) ----
  half_t* sO = (half_t*)sS;                 // reuse score buffer as [q][64] f16 staging
  {
    v8f o0 = {}, o1 = {};
#pragma unroll
    for (int k0 = 0; k0 < 64; k0 += 32) {
      v16h a  = load_frag(sP,  wmq * 16,      k0, 64, lane);
      v16h b0 = load_frag(sVT, wnk * 32,      k0, 64, lane);
      v16h b1 = load_frag(sVT, wnk * 32 + 16, k0, 64, lane);
      o0 = WMMA_F16(a, b0, o0);
      o1 = WMMA_F16(a, b1, o1);
    }
#pragma unroll
    for (int r = 0; r < 8; ++r) {
      const int q = wmq * 16 + r + rsub;
      sO[q * 64 + wnk * 32 + csub]      = (half_t)o0[r];
      sO[q * 64 + wnk * 32 + 16 + csub] = (half_t)o1[r];
    }
  }
  __syncthreads();

  // ---- coalesced b128 output stores ----
#pragma unroll
  for (int it = 0; it < 2; ++it) {          // 512 x v8h
    const int idx = tid + it * 256;
    const int row = idx >> 3, c8 = (idx & 7) * 8;
    const int wr = row >> 3, wc = row & 7;
    const int tok = (bl * 64 + nh_i * 8 + wr) * 64 + (nw_i * 8 + wc);
    *(v8h*)(outp + (size_t)tok * 512 + head * 64 + c8) = *(const v8h*)(sO + row * 64 + c8);
  }
}

extern "C" void kernel_launch(void* const* d_in, const int* in_sizes, int n_in,
                              void* d_out, int out_size, void* d_ws, size_t ws_size,
                              hipStream_t stream) {
  const float* x     = (const float*)d_in[0];   // [2,4,64,64,512] -> [32768,512]
  const float* w_qkv = (const float*)d_in[1];   // [512,1536]
  const float* pos   = (const float*)d_in[2];   // [15,15]
  const float* w_out = (const float*)d_in[3];   // [512,512]
  const float* b_out = (const float*)d_in[4];   // [512]
  float* out = (float*)d_out;                   // [32768,512]

  const int M = 32768, K = 512, N1 = 1536, N2 = 512;

  half_t* qkv_f16 = (half_t*)d_ws;                                   // 96 MB
  half_t* att_f16 = (half_t*)((char*)d_ws + (size_t)M * N1 * 2);     // +32 MB

  dim3 blk(256);

  // Phase 1: qkv projection (fp32 in -> f16 out)
  dim3 g1(N1 / 128, M / 64);
  gemm_wmma_kernel<false, false, true><<<g1, blk, 0, stream>>>(
      x, w_qkv, nullptr, qkv_f16, M, N1, K);

  // Phase 2: windowed attention (4096 = 8 bl * 8 heads * 64 windows)
  win_attn_kernel<<<dim3(4096), blk, 0, stream>>>(qkv_f16, pos, att_f16);

  // Phase 3: output projection (+bias, fp32 out)
  dim3 g3(N2 / 128, M / 64);
  gemm_wmma_kernel<true, true, false><<<g3, blk, 0, stream>>>(
      att_f16, w_out, b_out, out, M, N2, K);
}
